// Block_47888885351104
// MI455X (gfx1250) — matrix operand order repaired and checked by reading, not compiled, not run on hardware
//
#include <hip/hip_runtime.h>
#include <hip/hip_bf16.h>
#include <cstdint>
#include <cstddef>

// ---------------------------------------------------------------------------
// CDNA5 (gfx1250) transformer block: LN -> MHA (k@q^T, causal) -> proj+res ->
// LN -> FFN(ReLU)+res.  All matmuls via v_wmma_f32_16x16x32_bf16 (wave32).
// GEMM tiles staged with GLOBAL_LOAD_ASYNC_TO_LDS_B128 + ASYNCcnt waits,
// double-buffered in LDS.
// ---------------------------------------------------------------------------

typedef __attribute__((ext_vector_type(16))) __bf16 v16bf;
typedef __attribute__((ext_vector_type(8)))  float  v8f;

union FragBF {
  v16bf v;
  uint4 q[2];
  unsigned short h[16];
};

// ---- async global->LDS path (CDNA5), with sync fallback -------------------
#if defined(__has_builtin)
#if __has_builtin(__builtin_amdgcn_global_load_async_to_lds_b128) && \
    __has_builtin(__builtin_amdgcn_s_wait_asynccnt)
#define USE_ASYNC_LDS 1
#endif
#endif
#ifndef USE_ASYNC_LDS
#define USE_ASYNC_LDS 0
#endif

#if USE_ASYNC_LDS
typedef int v4i_b128 __attribute__((vector_size(16)));
typedef __attribute__((address_space(1))) v4i_b128* g_b128_ptr;
typedef __attribute__((address_space(3))) v4i_b128* l_b128_ptr;
#endif

__device__ __forceinline__ void copy16_g2l(const unsigned short* g,
                                           unsigned short* l) {
#if USE_ASYNC_LDS
  __builtin_amdgcn_global_load_async_to_lds_b128((g_b128_ptr)g, (l_b128_ptr)l,
                                                 0, 0);
#else
  *(uint4*)l = *(const uint4*)g;
#endif
}
__device__ __forceinline__ void async_wait_all() {
#if USE_ASYNC_LDS
  __builtin_amdgcn_s_wait_asynccnt(0);
#endif
}

__device__ __forceinline__ v8f vzero8() {
  v8f v;
#pragma unroll
  for (int i = 0; i < 8; ++i) v[i] = 0.0f;
  return v;
}

__device__ __forceinline__ unsigned short f2bf(float f) {
  unsigned int u = __float_as_uint(f);
  unsigned int r = u + 0x7fffu + ((u >> 16) & 1u);  // round-nearest-even
  return (unsigned short)(r >> 16);
}

// reductions confined to a 16-lane half of the wave32 (matches C-frag layout:
// lanes 0-15 hold rows i, lanes 16-31 hold rows i+8, col = lane%16)
__device__ __forceinline__ float half16_max(float v) {
#pragma unroll
  for (int d = 1; d < 16; d <<= 1) v = fmaxf(v, __shfl_xor(v, d, 32));
  return v;
}
__device__ __forceinline__ float half16_sum(float v) {
#pragma unroll
  for (int d = 1; d < 16; d <<= 1) v += __shfl_xor(v, d, 32);
  return v;
}

// ---------------------------------------------------------------------------
// LayerNorm: rows of 384, unbiased variance (ddof=1), gamma=1 beta=0.
// f32 in -> bf16 out.  One 128-thread block per row.
// ---------------------------------------------------------------------------
__global__ void __launch_bounds__(128) ln_bf16_kernel(
    const float* __restrict__ x, unsigned short* __restrict__ out) {
  const size_t row = blockIdx.x;
  const float* xr = x + row * 384;
  unsigned short* orow = out + row * 384;
  float v0 = xr[threadIdx.x];
  float v1 = xr[threadIdx.x + 128];
  float v2 = xr[threadIdx.x + 256];
  float s = v0 + v1 + v2;
  float q = v0 * v0 + v1 * v1 + v2 * v2;
#pragma unroll
  for (int d = 1; d < 32; d <<= 1) {
    s += __shfl_xor(s, d, 32);
    q += __shfl_xor(q, d, 32);
  }
  __shared__ float red[8];
  const int wid = threadIdx.x >> 5;
  if ((threadIdx.x & 31) == 0) { red[wid] = s; red[4 + wid] = q; }
  __syncthreads();
  s = red[0] + red[1] + red[2] + red[3];
  q = red[4] + red[5] + red[6] + red[7];
  const float mean = s * (1.0f / 384.0f);
  const float var  = (q - 384.0f * mean * mean) * (1.0f / 383.0f);
  const float rstd = rsqrtf(var + 1e-5f);
  orow[threadIdx.x]       = f2bf((v0 - mean) * rstd);
  orow[threadIdx.x + 128] = f2bf((v1 - mean) * rstd);
  orow[threadIdx.x + 256] = f2bf((v2 - mean) * rstd);
}

// ---------------------------------------------------------------------------
// Pack Wq|Wk|Wv ([H,C,HS] f32 each) into one bf16 [C, 1152] GEMM weight:
// cols [0,384)=Q, [384,768)=K, [768,1152)=V, each section h-major.
// ---------------------------------------------------------------------------
__global__ void pack_qkv_kernel(const float* __restrict__ Wq,
                                const float* __restrict__ Wk,
                                const float* __restrict__ Wv,
                                unsigned short* __restrict__ out) {
  const int idx = blockIdx.x * 256 + threadIdx.x;
  if (idx >= 384 * 1152) return;
  const int c = idx / 1152, j = idx % 1152;
  const int sec = j / 384, r = j % 384;
  const int h = r / 64, s = r % 64;
  const float* W = (sec == 0) ? Wq : ((sec == 1) ? Wk : Wv);
  out[idx] = f2bf(W[((size_t)h * 384 + c) * 64 + s]);
}

__global__ void cvt_bf16_kernel(const float* __restrict__ in,
                                unsigned short* __restrict__ out, int n) {
  const int i = blockIdx.x * 256 + threadIdx.x;
  if (i < n) out[i] = f2bf(in[i]);
}

// ---------------------------------------------------------------------------
// Generic bf16 WMMA GEMM: out[M,N] = A[M,K] @ B[K,N] (+bias)(ReLU)(+res f32)
// Block tile 128x64, BK=64, double-buffered LDS filled by async copies.
// 8 waves in 4(M)x2(N); wave tile 32x32 = 2x2 fragments.
// ---------------------------------------------------------------------------
#define GEMM_LDA 72   // 64 k + pad
#define GEMM_LDB 72   // 64 n + pad

__device__ __forceinline__ void gemm_stage_tiles(
    const unsigned short* __restrict__ A, const unsigned short* __restrict__ Bw,
    unsigned short* as, unsigned short* bs,
    int m0, int n0, int kt, int K, int N, int tid) {
  // A tile 128x64 = 1024 x 16B, B tile 64x64 = 512 x 16B; 256 threads
#pragma unroll
  for (int j = 0; j < 4; ++j) {
    const int id = j * 256 + tid;
    const int r = id >> 3, c = (id & 7) * 8;
    copy16_g2l(&A[(size_t)(m0 + r) * K + kt + c], &as[r * GEMM_LDA + c]);
  }
#pragma unroll
  for (int j = 0; j < 2; ++j) {
    const int id = j * 256 + tid;
    const int r = id >> 3, c = (id & 7) * 8;
    copy16_g2l(&Bw[(size_t)(kt + r) * N + n0 + c], &bs[r * GEMM_LDB + c]);
  }
}

__global__ void __launch_bounds__(256) gemm_bf16_kernel(
    const unsigned short* __restrict__ A, const unsigned short* __restrict__ Bw,
    const float* __restrict__ bias, const float* __restrict__ resid,
    unsigned short* __restrict__ outb, float* __restrict__ outf,
    int M, int N, int K, int relu) {
  (void)M;
  __shared__ unsigned short As[2][128 * GEMM_LDA];  // 2 x 18 KB
  __shared__ unsigned short Bs[2][64 * GEMM_LDB];   // 2 x 9 KB
  const int n0 = blockIdx.x * 64;
  const int m0 = blockIdx.y * 128;
  const int tid = threadIdx.x;
  const int lane = tid & 31;
  const int wave = tid >> 5;
  const int wm = wave & 3;             // 4 waves along M (32 rows each)
  const int wn = wave >> 2;            // 2 waves along N (32 cols each)
  const int hi = (lane & 16) ? 8 : 0;  // A-frag K sub-offset per ISA layout

  v8f acc[2][2];
#pragma unroll
  for (int a = 0; a < 2; ++a)
#pragma unroll
    for (int b = 0; b < 2; ++b) acc[a][b] = vzero8();

  // prologue: stage tile 0
  gemm_stage_tiles(A, Bw, As[0], Bs[0], m0, n0, 0, K, N, tid);
  async_wait_all();
  __syncthreads();

  const int ntiles = K >> 6;
  for (int t = 0; t < ntiles; ++t) {
    const int cur = t & 1;
    if (t + 1 < ntiles)  // async-prefetch next tile into the other buffer
      gemm_stage_tiles(A, Bw, As[cur ^ 1], Bs[cur ^ 1], m0, n0, (t + 1) << 6,
                       K, N, tid);
    const unsigned short* as = As[cur];
    const unsigned short* bs = Bs[cur];
#pragma unroll
    for (int ks = 0; ks < 2; ++ks) {
      FragBF af[2];
#pragma unroll
      for (int fm = 0; fm < 2; ++fm) {
        const unsigned short* p =
            &as[(wm * 32 + fm * 16 + (lane & 15)) * GEMM_LDA + ks * 32 + hi];
        af[fm].q[0] = *(const uint4*)p;
        af[fm].q[1] = *(const uint4*)(p + 16);
      }
#pragma unroll
      for (int fn = 0; fn < 2; ++fn) {
        FragBF bf;
        const unsigned short* p =
            &bs[(ks * 32 + hi) * GEMM_LDB + wn * 32 + fn * 16 + (lane & 15)];
#pragma unroll
        for (int e = 0; e < 16; ++e)
          bf.h[e] = p[(e + (e >= 8 ? 8 : 0)) * GEMM_LDB];
#pragma unroll
        for (int fm = 0; fm < 2; ++fm)
          acc[fm][fn] = __builtin_amdgcn_wmma_f32_16x16x32_bf16(
              false, af[fm].v, false, bf.v, (short)0, acc[fm][fn], false, false);
      }
    }
    async_wait_all();  // next-tile async copies landed
    __syncthreads();   // everyone done reading cur buffer
  }

  const int nl = lane & 15;
  const int roff = (lane & 16) ? 8 : 0;
#pragma unroll
  for (int fm = 0; fm < 2; ++fm)
#pragma unroll
    for (int fn = 0; fn < 2; ++fn)
#pragma unroll
      for (int i = 0; i < 8; ++i) {
        const int mg = m0 + wm * 32 + fm * 16 + i + roff;
        const int ng = n0 + wn * 32 + fn * 16 + nl;
        float v = acc[fm][fn][i];
        if (bias) v += bias[ng];
        if (relu) v = fmaxf(v, 0.0f);
        if (resid) v += resid[(size_t)mg * N + ng];
        if (outf) outf[(size_t)mg * N + ng] = v;
        else      outb[(size_t)mg * N + ng] = f2bf(v);
      }
}

// ---------------------------------------------------------------------------
// Fused causal attention (flash-style), matching the reference quirk:
//   S = K @ Q^T / sqrt(384), tril mask, softmax over u, O = P @ V.
// Block = (b, h, 128-row chunk of t); 8 waves x 16 rows each.
// qkv layout per row: [Q(h*64) | K(+384) | V(+768)] bf16, ld=1152.
// ---------------------------------------------------------------------------
__global__ void __launch_bounds__(256) attn_kernel(
    const unsigned short* __restrict__ qkv, unsigned short* __restrict__ attn) {
  __shared__ unsigned short Qt[64 * 264];        // Q^T: [s][u], padded ld=264
  __shared__ unsigned short Pb[8 * 16 * 40];     // per-wave P bounce tile
  const int b = blockIdx.z, h = blockIdx.y, tc = blockIdx.x;
  const int tid = threadIdx.x, lane = tid & 31, wave = tid >> 5;
  const size_t rowbase = (size_t)b * 256;

  // stage Q transposed: Q[u][s] -> Qt[s][u]
#pragma unroll
  for (int j = 0; j < 8; ++j) {
    const int id = j * 256 + tid;
    const int u = id >> 3, c = (id & 7) * 8;
    uint4 d = *(const uint4*)&qkv[(rowbase + u) * 1152 + h * 64 + c];
    const unsigned short* ds = (const unsigned short*)&d;
#pragma unroll
    for (int e = 0; e < 8; ++e) Qt[(c + e) * 264 + u] = ds[e];
  }
  __syncthreads();

  const int t_base = tc * 128 + wave * 16;
  const int nl = lane & 15;
  const int roff = (lane & 16) ? 8 : 0;
  const int hi = roff;  // A-frag K sub-offset

  // K rows (A matrix, 16x64) straight from global in ISA A-frag layout
  FragBF ka[2];
  {
    const unsigned short* kp =
        &qkv[(rowbase + t_base + nl) * 1152 + 384 + h * 64];
    ka[0].q[0] = *(const uint4*)(kp + hi);
    ka[0].q[1] = *(const uint4*)(kp + hi + 16);
    ka[1].q[0] = *(const uint4*)(kp + 32 + hi);
    ka[1].q[1] = *(const uint4*)(kp + 32 + hi + 16);
  }

  v8f O[4];
#pragma unroll
  for (int f = 0; f < 4; ++f) O[f] = vzero8();
  float mrun[8], lrun[8];
#pragma unroll
  for (int i = 0; i < 8; ++i) { mrun[i] = -__builtin_inff(); lrun[i] = 0.0f; }
  const float scale = 0.05103103630798287f;  // 1/sqrt(384)
  unsigned short* pb = &Pb[wave * 16 * 40];

  for (int u0 = 0; u0 <= t_base + 15; u0 += 32) {
    // S(16x32) = K(16x64) @ Q^T(64x32) via 2 k-steps x 2 n-halves
    v8f s0 = vzero8(), s1 = vzero8();
#pragma unroll
    for (int j = 0; j < 2; ++j) {
      const unsigned short* qr = &Qt[(j * 32 + hi) * 264 + u0 + nl];
      FragBF b0, b1;
#pragma unroll
      for (int e = 0; e < 16; ++e) {
        b0.h[e] = qr[(e + (e >= 8 ? 8 : 0)) * 264];
        b1.h[e] = qr[(e + (e >= 8 ? 8 : 0)) * 264 + 16];
      }
      s0 = __builtin_amdgcn_wmma_f32_16x16x32_bf16(false, ka[j].v, false, b0.v,
                                                   (short)0, s0, false, false);
      s1 = __builtin_amdgcn_wmma_f32_16x16x32_bf16(false, ka[j].v, false, b1.v,
                                                   (short)0, s1, false, false);
    }
    // scale, causal mask, online softmax (per-row state aligned to frag rows)
    float alpha[8];
#pragma unroll
    for (int i = 0; i < 8; ++i) {
      const int t = t_base + i + roff;
      float a = s0[i] * scale; a = (u0 + nl       <= t) ? a : -__builtin_inff();
      float c = s1[i] * scale; c = (u0 + 16 + nl  <= t) ? c : -__builtin_inff();
      const float mi = half16_max(fmaxf(a, c));
      const float mn = fmaxf(mrun[i], mi);
      const float al = __expf(mrun[i] - mn);
      const float p0 = __expf(a - mn);
      const float p1 = __expf(c - mn);
      const float rs = half16_sum(p0 + p1);
      lrun[i] = lrun[i] * al + rs;
      mrun[i] = mn;
      alpha[i] = al;
      pb[(i + roff) * 40 + nl]      = f2bf(p0);
      pb[(i + roff) * 40 + 16 + nl] = f2bf(p1);
    }
#pragma unroll
    for (int f = 0; f < 4; ++f)
#pragma unroll
      for (int i = 0; i < 8; ++i) O[f][i] *= alpha[i];
    // P back out of LDS in A-frag layout (wave-local, DS ops are in-order)
    FragBF pa;
    {
      const unsigned short* pr = pb + nl * 40 + hi;
      pa.q[0] = *(const uint4*)pr;
      pa.q[1] = *(const uint4*)(pr + 16);
    }
    // V rows u0..u0+31 straight from global in B-frag layout; O += P@V
    const unsigned short* vp = &qkv[(rowbase + u0 + hi) * 1152 + 768 + h * 64 + nl];
#pragma unroll
    for (int f = 0; f < 4; ++f) {
      FragBF vb;
#pragma unroll
      for (int e = 0; e < 16; ++e)
        vb.h[e] = vp[(size_t)(e + (e >= 8 ? 8 : 0)) * 1152 + f * 16];
      O[f] = __builtin_amdgcn_wmma_f32_16x16x32_bf16(false, pa.v, false, vb.v,
                                                     (short)0, O[f], false, false);
    }
  }

#pragma unroll
  for (int i = 0; i < 8; ++i) lrun[i] = 1.0f / lrun[i];
#pragma unroll
  for (int f = 0; f < 4; ++f)
#pragma unroll
    for (int i = 0; i < 8; ++i)
      attn[(rowbase + t_base + i + roff) * 384 + h * 64 + f * 16 + nl] =
          f2bf(O[f][i] * lrun[i]);
}

// ---------------------------------------------------------------------------
// Host orchestration (graph-capture safe: only launches on `stream`).
// Inputs: x, Wk, Wq, Wv, Wo, bo, W1, b1, W2, b2.  Output f32 [B,T,C].
// ---------------------------------------------------------------------------
extern "C" void kernel_launch(void* const* d_in, const int* in_sizes, int n_in,
                              void* d_out, int out_size, void* d_ws,
                              size_t ws_size, hipStream_t stream) {
  (void)in_sizes; (void)n_in; (void)out_size; (void)ws_size;
  const float* x  = (const float*)d_in[0];
  const float* Wk = (const float*)d_in[1];
  const float* Wq = (const float*)d_in[2];
  const float* Wv = (const float*)d_in[3];
  const float* Wo = (const float*)d_in[4];
  const float* bo = (const float*)d_in[5];
  const float* W1 = (const float*)d_in[6];
  const float* b1 = (const float*)d_in[7];
  const float* W2 = (const float*)d_in[8];
  const float* b2 = (const float*)d_in[9];
  float* out = (float*)d_out;

  const int BT = 256 * 256, C = 384, C4 = 1536, NQKV = 1152;
  char* ws = (char*)d_ws;
  size_t off = 0;
  auto take = [&](size_t bytes) -> char* {
    char* p = ws + off;
    off = (off + bytes + 255) & ~(size_t)255;
    return p;
  };
  unsigned short* hbf  = (unsigned short*)take((size_t)BT * C * 2);    // LN1 out, reused as attn out
  unsigned short* qkv  = (unsigned short*)take((size_t)BT * NQKV * 2); // reused as h2
  float*          x1   = (float*)take((size_t)BT * C * 4);             // x + attn proj
  unsigned short* ff1  = (unsigned short*)take((size_t)BT * C4 * 2);
  unsigned short* wqkv = (unsigned short*)take((size_t)C * NQKV * 2);
  unsigned short* wo   = (unsigned short*)take((size_t)C * C * 2);
  unsigned short* w1   = (unsigned short*)take((size_t)C * C4 * 2);
  unsigned short* w2   = (unsigned short*)take((size_t)C4 * C * 2);

  pack_qkv_kernel<<<(C * NQKV + 255) / 256, 256, 0, stream>>>(Wq, Wk, Wv, wqkv);
  cvt_bf16_kernel<<<(C * C + 255) / 256, 256, 0, stream>>>(Wo, wo, C * C);
  cvt_bf16_kernel<<<(C * C4 + 255) / 256, 256, 0, stream>>>(W1, w1, C * C4);
  cvt_bf16_kernel<<<(C * C4 + 255) / 256, 256, 0, stream>>>(W2, w2, C4 * C);

  ln_bf16_kernel<<<BT, 128, 0, stream>>>(x, hbf);
  gemm_bf16_kernel<<<dim3(NQKV / 64, BT / 128), 256, 0, stream>>>(
      hbf, wqkv, nullptr, nullptr, qkv, nullptr, BT, NQKV, C, 0);
  attn_kernel<<<dim3(2, 6, 256), 256, 0, stream>>>(qkv, hbf);
  gemm_bf16_kernel<<<dim3(C / 64, BT / 128), 256, 0, stream>>>(
      hbf, wo, bo, x, nullptr, x1, BT, C, C, 0);
  ln_bf16_kernel<<<BT, 128, 0, stream>>>(x1, qkv);  // h2 reuses qkv buffer
  gemm_bf16_kernel<<<dim3(C4 / 64, BT / 128), 256, 0, stream>>>(
      qkv, w1, b1, nullptr, ff1, nullptr, BT, C4, C, 1);
  gemm_bf16_kernel<<<dim3(C / 64, BT / 128), 256, 0, stream>>>(
      ff1, w2, b2, x1, nullptr, out, BT, C, C4, 0);
}
